// MultiHeadAttention_73014444032803
// MI455X (gfx1250) — compile-verified, matrix operand order checked by reading, not run on hardware
//
#include <hip/hip_runtime.h>
#include <hip/hip_bf16.h>

// ---------------------------------------------------------------------------
// Multi-head attention forward for MI455X (gfx1250, wave32, WMMA + TDM).
// B=2, S=2048, D_MODEL=768, H=12, HD=64.  All matmuls via
// v_wmma_f32_16x16x32_bf16; GEMM weight tiles staged to LDS by the Tensor
// Data Mover (tensor_load_to_lds, TENSORcnt double-buffering).
// ---------------------------------------------------------------------------

typedef unsigned int  u32;
typedef unsigned short u16;
typedef unsigned long long u64;
typedef __attribute__((ext_vector_type(16))) __bf16 v16bf;
typedef __attribute__((ext_vector_type(8)))  float  v8f;
typedef __attribute__((ext_vector_type(4)))  u32    v4u;
typedef __attribute__((ext_vector_type(8)))  u32    v8u;

#define D_MODEL 768
#define NHEADS  12
#define HDIM    64
#define BATCH   2
#define SEQ     2048
#define MROWS   (BATCH * SEQ)          // 4096
#define SCALE   0.125f                 // 64^-0.5

__device__ __forceinline__ u16 f2bf(float f) {
    u32 u = __float_as_uint(f);
    u += 0x7FFFu + ((u >> 16) & 1u);   // round-to-nearest-even
    return (u16)(u >> 16);
}

__device__ __forceinline__ v8f zero8() {
    v8f z;
#pragma unroll
    for (int e = 0; e < 8; ++e) z[e] = 0.0f;
    return z;
}

// Assemble a 16x32 bf16 WMMA A/B fragment for this lane from two contiguous
// 16-byte chunks at p and p+16 elements (K {0..7,16..23} / {8..15,24..31}
// per-half-wave striping of the 16-bit operand layout).
__device__ __forceinline__ v16bf load_frag(const u16* p) {
    v4u lo = *(const v4u*)p;
    v4u hi = *(const v4u*)(p + 16);
    v8u u;
    u[0] = lo[0]; u[1] = lo[1]; u[2] = lo[2]; u[3] = lo[3];
    u[4] = hi[0]; u[5] = hi[1]; u[6] = hi[2]; u[7] = hi[3];
    return __builtin_bit_cast(v16bf, u);
}

__device__ __forceinline__ v8f wmma_bf16(v16bf a, v16bf b, v8f c) {
    return __builtin_amdgcn_wmma_f32_16x16x32_bf16(
        false, a, false, b, (short)0, c, false, false);
}

// ---------------------------------------------------------------------------
// TDM: DMA a 64-row x 32-col bf16 tile of W (row-major, pitch K elems) into
// LDS.  2-D tensor -> 2 descriptor groups (VADDR2/3 = NULL form).
// All operands are wave-uniform scalars.
// ---------------------------------------------------------------------------
__device__ __forceinline__ void tdm_load_w_tile(const u16* gsrc, const u16* ldst,
                                                int K, int N) {
    u64 ga = (u64)(uintptr_t)gsrc;
    u32 lds_addr = (u32)(uintptr_t)ldst;       // generic addr[31:0] == LDS offset
    v4u g0;
    g0[0] = 1u;                                 // count=1, user descriptor
    g0[1] = lds_addr;                           // D#.lds_addr
    g0[2] = (u32)ga;                            // D#.global_addr[31:0]
    g0[3] = (u32)(ga >> 32) | 0x80000000u;      // global_addr[56:32] | type=2
    v8u g1;
    g1[0] = 0x00010000u;                        // wg_mask=0, data_size=2B
    g1[1] = ((u32)K & 0xFFFFu) << 16;           // tensor_dim0[15:0]
    g1[2] = ((u32)N & 0xFFFFu) << 16;           // dim0[31:16]=0 | tensor_dim1[15:0]
    g1[3] = 32u << 16;                          // dim1[31:16]=0 | tile_dim0=32
    g1[4] = 64u;                                // tile_dim1=64, tile_dim2=0
    g1[5] = (u32)K;                             // tensor_dim0_stride[31:0]
    g1[6] = 0u;                                 // stride hi | dim1_stride lo
    g1[7] = 0u;
    asm volatile("tensor_load_to_lds %0, %1" :: "s"(g0), "s"(g1) : "memory");
}

// ---------------------------------------------------------------------------
// fp32 -> bf16 conversion (vectorized x4)
// ---------------------------------------------------------------------------
__global__ void cvt_f32_bf16(const float* __restrict__ src,
                             u16* __restrict__ dst, int n4) {
    int i = blockIdx.x * blockDim.x + threadIdx.x;
    if (i < n4) {
        float4 f = ((const float4*)src)[i];
        ushort4 o;
        o.x = f2bf(f.x); o.y = f2bf(f.y); o.z = f2bf(f.z); o.w = f2bf(f.w);
        ((ushort4*)dst)[i] = o;
    }
}

// ---------------------------------------------------------------------------
// GEMM: C[M,N] = A[M,K] (bf16, row-major) @ W[N,K]^T (bf16, row-major rows=N)
// mode 0: store f32 row-major [M,N]            (final output projection)
// mode 1: store bf16 to [b,h,s,d]              (Q, K heads layout)
// mode 2: store bf16 to [b,h,d,s] (transposed) (V for the P*V matmul)
// Workgroup: 256 threads = 8 waves, WG tile 128x128, wave tile 32x64.
// B (weight) tiles staged via TDM into per-wave double-buffered LDS.
// Steady-state loop is branch-free; last K-step peeled into an epilogue.
// ---------------------------------------------------------------------------
__global__ __launch_bounds__(256)
void gemm_bf16(const u16* __restrict__ A, const u16* __restrict__ W,
               float* __restrict__ outf, u16* __restrict__ outb,
               int M, int N, int K, int mode) {
    __shared__ u16 ldsW[8 * 2 * 64 * 32];      // 64 KB: per wave 2 x (64x32) bufs

    const int lane = threadIdx.x & 31;
    const int w    = threadIdx.x >> 5;
    const int l16  = lane & 15;
    const int lh   = lane >> 4;
    const int m0   = blockIdx.y * 128 + (w & 3) * 32;
    const int n0   = blockIdx.x * 128 + (w >> 2) * 64;

    u16* myB = ldsW + w * (2 * 2048);          // wave-private LDS slice

    v8f c[2][4];
#pragma unroll
    for (int i = 0; i < 2; ++i)
#pragma unroll
        for (int j = 0; j < 4; ++j) c[i][j] = zero8();

    const u16* arow0 = A + (size_t)(m0 +      l16) * K + lh * 8;
    const u16* arow1 = A + (size_t)(m0 + 16 + l16) * K + lh * 8;
    const u16* wbase = W + (size_t)n0 * K;

    auto compute_step = [&](int kk, int bsel) {
        __builtin_prefetch(arow0 + kk + 64, 0, 1);
        __builtin_prefetch(arow1 + kk + 64, 0, 1);
        v16bf a0 = load_frag(arow0 + kk);
        v16bf a1 = load_frag(arow1 + kk);
        const u16* bb = myB + bsel * 2048;      // 64 rows x 32 cols, row-major
#pragma unroll
        for (int j = 0; j < 4; ++j) {
            v16bf b = load_frag(bb + (j * 16 + l16) * 32 + lh * 8);
            c[0][j] = wmma_bf16(a0, b, c[0][j]);
            c[1][j] = wmma_bf16(a1, b, c[1][j]);
        }
    };

    // prologue: DMA first weight tile
    tdm_load_w_tile(wbase, myB, K, N);

    int buf = 0;
    int k0 = 0;
    for (; k0 + 32 < K; k0 += 32) {             // branch-free steady state
        tdm_load_w_tile(wbase + k0 + 32, myB + (buf ^ 1) * 2048, K, N);
        asm volatile("s_wait_tensorcnt 0x1" ::: "memory"); // older (cur) tile done
        compute_step(k0, buf);
        buf ^= 1;
    }
    // epilogue: last K-step
    asm volatile("s_wait_tensorcnt 0x0" ::: "memory");
    compute_step(k0, buf);

    // C layout: VGPR r -> row (r + 8*lh) of 16x16 tile, lane%16 -> column.
#pragma unroll
    for (int i = 0; i < 2; ++i) {
#pragma unroll
        for (int j = 0; j < 4; ++j) {
            const int coln = n0 + j * 16 + l16;
#pragma unroll
            for (int r = 0; r < 8; ++r) {
                const int m = m0 + i * 16 + r + 8 * lh;
                const float v = c[i][j][r];
                if (mode == 0) {
                    outf[(size_t)m * N + coln] = v;
                } else {
                    const int b = m >> 11;          // m / SEQ
                    const int s = m & (SEQ - 1);
                    const int h = coln >> 6;        // coln / HDIM
                    const int d = coln & (HDIM - 1);
                    size_t dst = (mode == 1)
                        ? ((size_t)((b * NHEADS + h) * SEQ + s) * HDIM + d)
                        : ((size_t)((b * NHEADS + h) * HDIM + d) * SEQ + s);
                    outb[dst] = f2bf(v);
                }
            }
        }
    }
}

// ---------------------------------------------------------------------------
// Causal flash attention: one wave handles one (b,h, 16-query-row) tile.
// Online softmax in registers; P transits LDS (C-frag -> A-frag relayout).
// Qh,Kh: [b,h,s,d] bf16;  Vt: [b,h,d,s] bf16;  ctx: [b*S, D_MODEL] bf16.
// ---------------------------------------------------------------------------
__global__ __launch_bounds__(256)
void attn_flash(const u16* __restrict__ Qh, const u16* __restrict__ Kh,
                const u16* __restrict__ Vt, u16* __restrict__ ctx) {
    __shared__ u16 ptile[8 * 16 * 32];       // 1 KB per wave

    const int lane = threadIdx.x & 31;
    const int w    = threadIdx.x >> 5;
    const int l16  = lane & 15;
    const int lh   = lane >> 4;

    const int wid   = blockIdx.x * 8 + w;    // 0 .. 3071
    const int bh    = wid >> 7;              // / (SEQ/16)
    const int qt    = wid & 127;
    const int qbase = qt * 16;
    const int b     = bh / NHEADS;
    const int h     = bh % NHEADS;

    const u16* Qb = Qh + (size_t)bh * SEQ * HDIM;
    const u16* Kb = Kh + (size_t)bh * SEQ * HDIM;
    const u16* Vb = Vt + (size_t)bh * HDIM * SEQ;
    u16* lds = ptile + w * 512;

    const u16* qrow = Qb + (size_t)(qbase + l16) * HDIM + lh * 8;
    const v16bf qa0 = load_frag(qrow);        // d 0..31
    const v16bf qa1 = load_frag(qrow + 32);   // d 32..63

    v8f o[4];
#pragma unroll
    for (int dt = 0; dt < 4; ++dt) o[dt] = zero8();
    float mrow[8], lrow[8];
#pragma unroll
    for (int r = 0; r < 8; ++r) { mrow[r] = -1e30f; lrow[r] = 0.0f; }

    const int jblocks = (qbase + 15) / 32 + 1;   // causal bound, 32 keys/iter
    for (int jb = 0; jb < jblocks; ++jb) {
        const int kb0 = jb * 32;

        v8f s[2];
#pragma unroll
        for (int t = 0; t < 2; ++t) {
            s[t] = zero8();
            const u16* krow = Kb + (size_t)(kb0 + t * 16 + l16) * HDIM + lh * 8;
            s[t] = wmma_bf16(qa0, load_frag(krow),      s[t]);
            s[t] = wmma_bf16(qa1, load_frag(krow + 32), s[t]);
        }

#pragma unroll
        for (int t = 0; t < 2; ++t) {
            const int kidx = kb0 + t * 16 + l16;
#pragma unroll
            for (int r = 0; r < 8; ++r) {
                const int qidx = qbase + r + 8 * lh;
                const float v = s[t][r] * SCALE;
                s[t][r] = (kidx > qidx) ? -1e30f : v;
            }
        }

        float scl[8];
#pragma unroll
        for (int r = 0; r < 8; ++r) {
            float vmax = fmaxf(s[0][r], s[1][r]);
            vmax = fmaxf(vmax, __shfl_xor(vmax, 1));
            vmax = fmaxf(vmax, __shfl_xor(vmax, 2));
            vmax = fmaxf(vmax, __shfl_xor(vmax, 4));
            vmax = fmaxf(vmax, __shfl_xor(vmax, 8));
            const float mnew = fmaxf(mrow[r], vmax);
            scl[r] = __expf(mrow[r] - mnew);
            const float p0 = __expf(s[0][r] - mnew);
            const float p1 = __expf(s[1][r] - mnew);
            s[0][r] = p0; s[1][r] = p1;
            float rs = p0 + p1;
            rs += __shfl_xor(rs, 1);
            rs += __shfl_xor(rs, 2);
            rs += __shfl_xor(rs, 4);
            rs += __shfl_xor(rs, 8);
            lrow[r] = lrow[r] * scl[r] + rs;
            mrow[r] = mnew;
        }

#pragma unroll
        for (int dt = 0; dt < 4; ++dt)
#pragma unroll
            for (int r = 0; r < 8; ++r) o[dt][r] *= scl[r];

        // P (16x32, C-layout) -> LDS -> A-fragment relayout (in-order DS,
        // wave-private slice: no barrier needed, loop lengths diverge per wave)
#pragma unroll
        for (int t = 0; t < 2; ++t)
#pragma unroll
            for (int r = 0; r < 8; ++r) {
                const int row = r + 8 * lh;
                const int col = t * 16 + l16;
                lds[row * 32 + col] = f2bf(s[t][r]);
            }
        const v16bf pa = load_frag(lds + l16 * 32 + lh * 8);

#pragma unroll
        for (int dt = 0; dt < 4; ++dt) {
            const u16* vrow = Vb + (size_t)(dt * 16 + l16) * SEQ + kb0 + lh * 8;
            o[dt] = wmma_bf16(pa, load_frag(vrow), o[dt]);
        }
    }

    float inv[8];
#pragma unroll
    for (int r = 0; r < 8; ++r) inv[r] = 1.0f / lrow[r];
#pragma unroll
    for (int dt = 0; dt < 4; ++dt) {
        const int col = h * HDIM + dt * 16 + l16;
#pragma unroll
        for (int r = 0; r < 8; ++r) {
            const int m = b * SEQ + qbase + r + 8 * lh;
            ctx[(size_t)m * D_MODEL + col] = f2bf(o[dt][r] * inv[r]);
        }
    }
}

// ---------------------------------------------------------------------------
// Host-side launch
// ---------------------------------------------------------------------------
extern "C" void kernel_launch(void* const* d_in, const int* in_sizes, int n_in,
                              void* d_out, int out_size, void* d_ws, size_t ws_size,
                              hipStream_t stream) {
    (void)in_sizes; (void)n_in; (void)out_size; (void)ws_size;

    const float* q_in = (const float*)d_in[0];
    const float* k_in = (const float*)d_in[1];
    const float* v_in = (const float*)d_in[2];
    const float* Wq   = (const float*)d_in[3];
    const float* Wk   = (const float*)d_in[5];
    const float* Wv   = (const float*)d_in[7];
    const float* Wo   = (const float*)d_in[9];
    float* out = (float*)d_out;

    const size_t NE = (size_t)MROWS * D_MODEL;    // 3,145,728 activations
    const size_t WE = (size_t)D_MODEL * D_MODEL;  //   589,824 weights

    u16* xq  = (u16*)d_ws;
    u16* xk  = xq  + NE;
    u16* xv  = xk  + NE;
    u16* wq  = xv  + NE;
    u16* wk  = wq  + WE;
    u16* wv  = wk  + WE;
    u16* wo  = wv  + WE;
    u16* Qh  = wo  + WE;      // [b,h,s,d]
    u16* Kh  = Qh  + NE;      // [b,h,s,d]
    u16* Vt  = Kh  + NE;      // [b,h,d,s]
    u16* ctx = Vt  + NE;      // [m, 768]

    {
        const int na4 = (int)(NE / 4), nw4 = (int)(WE / 4);
        const int ba = (na4 + 255) / 256, bw = (nw4 + 255) / 256;
        cvt_f32_bf16<<<ba, 256, 0, stream>>>(q_in, xq, na4);
        cvt_f32_bf16<<<ba, 256, 0, stream>>>(k_in, xk, na4);
        cvt_f32_bf16<<<ba, 256, 0, stream>>>(v_in, xv, na4);
        cvt_f32_bf16<<<bw, 256, 0, stream>>>(Wq, wq, nw4);
        cvt_f32_bf16<<<bw, 256, 0, stream>>>(Wk, wk, nw4);
        cvt_f32_bf16<<<bw, 256, 0, stream>>>(Wv, wv, nw4);
        cvt_f32_bf16<<<bw, 256, 0, stream>>>(Wo, wo, nw4);
    }

    dim3 gg(D_MODEL / 128, MROWS / 128);   // (6, 32)
    gemm_bf16<<<gg, 256, 0, stream>>>(xq, wq, nullptr, Qh, MROWS, D_MODEL, D_MODEL, 1);
    gemm_bf16<<<gg, 256, 0, stream>>>(xk, wk, nullptr, Kh, MROWS, D_MODEL, D_MODEL, 1);
    gemm_bf16<<<gg, 256, 0, stream>>>(xv, wv, nullptr, Vt, MROWS, D_MODEL, D_MODEL, 2);

    attn_flash<<<384, 256, 0, stream>>>(Qh, Kh, Vt, ctx);

    gemm_bf16<<<gg, 256, 0, stream>>>(ctx, wo, out, nullptr, MROWS, D_MODEL, D_MODEL, 0);
}
